// Kistmat_AI_51196010168796
// MI455X (gfx1250) — compile-verified, matrix operand order checked by reading, not run on hardware
//
#include <hip/hip_runtime.h>
#include <hip/hip_bf16.h>
#include <math.h>

// ---------------------------------------------------------------------------
// MI455X (gfx1250) implementation.
// Heavy path: persistent wave32 WMMA LSTM kernels (v_wmma_f32_16x16x32_f16).
// Weights are pre-swizzled (f32->f16, WMMA B-operand lane order) into the
// workspace once, then each block pulls its slice into LDS with
// GLOBAL_LOAD_ASYNC_TO_LDS_B128 (ASYNCcnt path) + s_wait_asynccnt.
// Per-step inter-WGP barrier via global atomics. Attention over seq-len 1
// collapses to v@Wo (softmax of one logit == 1), so Wq/Wk are skipped.
// ---------------------------------------------------------------------------

typedef __attribute__((ext_vector_type(16))) _Float16 v16h;
typedef __attribute__((ext_vector_type(8)))  _Float16 v8h;
typedef __attribute__((ext_vector_type(8)))  float    v8f;

__device__ __forceinline__ float sigmf(float x) { return 1.0f / (1.0f + expf(-x)); }

// ---------------------------------------------------------------------------
// Embedding gather + f32 -> f16
// ---------------------------------------------------------------------------
__global__ void embed_f16_kernel(const int* __restrict__ tokens,
                                 const float* __restrict__ emb,
                                 _Float16* __restrict__ x0, int n) {
  int i = blockIdx.x * blockDim.x + threadIdx.x;
  if (i >= n) return;
  int tok = tokens[i >> 6];
  x0[i] = (_Float16)emb[(size_t)tok * 64 + (i & 63)];
}

// ---------------------------------------------------------------------------
// Pre-swizzle [Wx;Wh] (f32) into the WMMA-B-ready f16 image used by the LSTM
// kernel: dst[dir][nb][kb][q][lane][16]. Element (lane,i) of a B fragment is
// W[k = kb*32 + klocal(lane,i)][col = q*512 + nb*16 + (lane&15)].
// ---------------------------------------------------------------------------
__global__ void swizzle_weights_kernel(
    const float* __restrict__ Wx_f, const float* __restrict__ Wh_f,
    const float* __restrict__ Wx_b, const float* __restrict__ Wh_b,
    _Float16* __restrict__ dst, int Din, int total) {
  int gi = blockIdx.x * blockDim.x + threadIdx.x;
  if (gi >= total) return;
  const int chunk = total >> 6;           // NKB*2048 elems per (dir,nb)
  int dirnb = gi / chunk;
  int idx   = gi - dirnb * chunk;
  int dir = dirnb >> 5, nb = dirnb & 31;
  int i  = idx & 15;
  int ln = (idx >> 4) & 31;
  int q  = (idx >> 9) & 3;
  int kb = idx >> 11;
  int klocal = ((i >> 3) << 4) | ((ln >> 4) << 3) | (i & 7);
  int k   = kb * 32 + klocal;
  int col = q * 512 + nb * 16 + (ln & 15);
  const float* Wx = dir ? Wx_b : Wx_f;
  const float* Wh = dir ? Wh_b : Wh_f;
  float w = (k < Din) ? Wx[(size_t)k * 2048 + col]
                      : Wh[(size_t)(k - Din) * 2048 + col];
  dst[gi] = (_Float16)w;
}

// ---------------------------------------------------------------------------
// Persistent BiLSTM layer kernel.
// Grid: 64 blocks x 256 threads. blockIdx>>5 = direction, blockIdx&31 = nb
// (owns H columns [16*nb, 16*nb+16) of all four gates i,f,g,o).
// gates[64, 4x16] = [x_t ; h] (f16) @ swizzled weight slice (f16 in LDS).
// A layout (16x32 f16): lanes 0-15 row M=lane, halves K 0..7,16..23;
// lanes 16-31 same row, K 8..15,24..31. B mirrors it per column.
// C/D f32 16x16: lane n holds col n, rows rb..rb+7 (rb = 8*(lane>=16)).
// ---------------------------------------------------------------------------
template <int DIN>
__global__ void lstm_layer_kernel(
    const _Float16* __restrict__ x,          // [64, T, DIN] f16
    const _Float16* __restrict__ wswz,       // [2][32][NKB*2048] f16 swizzled
    const float* __restrict__ b_f, const float* __restrict__ b_b,
    _Float16* __restrict__ hbuf,             // [2][64][512] f16 (zeroed)
    float*    __restrict__ cbuf,             // [2][64][512] f32 (zeroed)
    _Float16* __restrict__ yout,             // seq [64,T,1024] or last [64,1024]
    int T, int writeSeq,
    int* __restrict__ barrier_cnt)           // [2] ints (zeroed)
{
  constexpr int KTOT = DIN + 512;
  constexpr int NKB  = KTOT / 32;

  extern __shared__ _Float16 smem[];
  _Float16* bsw    = smem;                        // [NKB][4][32][16] f16
  float*    gatesL = (float*)(smem + NKB * 2048); // [4][64][16] f32

  const int tid = threadIdx.x;
  const int dir = blockIdx.x >> 5;
  const int nb  = blockIdx.x & 31;

  const float* bias = dir ? b_b : b_f;
  _Float16* hdir = hbuf + (size_t)dir * 64 * 512;
  float*    cdir = cbuf + (size_t)dir * 64 * 512;

  // ---- async-copy this block's pre-swizzled weight slice into LDS.
  // Per-lane LDS dst + per-lane global src, 16B per op: the exact addressing
  // model of GLOBAL_LOAD_ASYNC_TO_LDS_B128 (tracked on ASYNCcnt).
  {
    const _Float16* wchunk = wswz + (size_t)(dir * 32 + nb) * (NKB * 2048);
    const int nchunks = NKB * 256;               // 16-byte chunks
    for (int ci = tid; ci < nchunks; ci += blockDim.x) {
      uint32_t lp = (uint32_t)(uintptr_t)bsw + (uint32_t)(ci * 16);
      uint64_t gp = (uint64_t)(uintptr_t)wchunk + (uint64_t)(ci * 16);
      asm volatile("global_load_async_to_lds_b128 %0, %1, off"
                   :: "v"(lp), "v"(gp) : "memory");
    }
    asm volatile("s_wait_asynccnt 0x0" ::: "memory");
  }
  __syncthreads();

  const int wave = tid >> 5;
  const int lane = tid & 31;
  const int m    = wave & 3;             // M tile (batch rows 16m..16m+15)
  const int q0   = (wave >> 2) * 2;      // two gate tiles per wave
  const int q1   = q0 + 1;
  const int arow = m * 16 + (lane & 15); // batch row this lane feeds for A
  const int kb8  = (lane >> 4) << 3;     // 0 or 8 (A/B half-of-K select)

  for (int s = 0; s < T; ++s) {
    const int t = dir ? (T - 1 - s) : s;

    v8f acc0 = {};
    v8f acc1 = {};
    for (int kb = 0; kb < NKB; ++kb) {
      const int k0 = kb * 32;
      // ---- A fragment: rows of [x_t ; h]
      const _Float16* pa = (k0 < DIN)
          ? (x + ((size_t)arow * T + t) * DIN + k0 + kb8)
          : (hdir + (size_t)arow * 512 + (k0 - DIN) + kb8);
      v8h alo = *(const v8h*)pa;
      v8h ahi = *(const v8h*)(pa + 16);
      v16h a;
#pragma unroll
      for (int i = 0; i < 8; ++i) { a[i] = alo[i]; a[8 + i] = ahi[i]; }

      // ---- B fragments from pre-swizzled LDS (contiguous per lane)
      const _Float16* pb0 = &bsw[((kb * 4 + q0) * 32 + lane) * 16];
      const _Float16* pb1 = &bsw[((kb * 4 + q1) * 32 + lane) * 16];
      v8h b0lo = *(const v8h*)pb0; v8h b0hi = *(const v8h*)(pb0 + 8);
      v8h b1lo = *(const v8h*)pb1; v8h b1hi = *(const v8h*)(pb1 + 8);
      v16h b0, b1;
#pragma unroll
      for (int i = 0; i < 8; ++i) {
        b0[i] = b0lo[i]; b0[8 + i] = b0hi[i];
        b1[i] = b1lo[i]; b1[8 + i] = b1hi[i];
      }

      acc0 = __builtin_amdgcn_wmma_f32_16x16x32_f16(false, a, false, b0,
                                                    (short)0, acc0, false, false);
      acc1 = __builtin_amdgcn_wmma_f32_16x16x32_f16(false, a, false, b1,
                                                    (short)0, acc1, false, false);
    }

    // ---- spill gate tiles to LDS (C layout: lane&15 = col, rows rb..rb+7)
    {
      const int col = lane & 15;
      const int rb  = (lane >> 4) * 8;
#pragma unroll
      for (int r = 0; r < 8; ++r) {
        gatesL[(q0 * 64 + m * 16 + rb + r) * 16 + col] = acc0[r];
        gatesL[(q1 * 64 + m * 16 + rb + r) * 16 + col] = acc1[r];
      }
    }
    __syncthreads();

    // ---- LSTM cell update for this block's 16 H columns (Keras i,f,g,o)
    for (int e = tid; e < 64 * 16; e += blockDim.x) {
      const int row  = e >> 4;
      const int cc   = e & 15;
      const int hcol = nb * 16 + cc;
      float gi = gatesL[(0 * 64 + row) * 16 + cc] + bias[hcol];
      float gf = gatesL[(1 * 64 + row) * 16 + cc] + bias[512 + hcol];
      float gg = gatesL[(2 * 64 + row) * 16 + cc] + bias[1024 + hcol];
      float go = gatesL[(3 * 64 + row) * 16 + cc] + bias[1536 + hcol];
      float c  = cdir[(size_t)row * 512 + hcol];
      c = sigmf(gf) * c + sigmf(gi) * tanhf(gg);
      float h = sigmf(go) * tanhf(c);
      cdir[(size_t)row * 512 + hcol] = c;
      hdir[(size_t)row * 512 + hcol] = (_Float16)h;
      if (writeSeq) {
        yout[((size_t)row * T + t) * 1024 + (size_t)dir * 512 + hcol] = (_Float16)h;
      } else if (s == T - 1) {
        yout[(size_t)row * 1024 + dir * 512 + hcol] = (_Float16)h;
      }
    }

    // ---- prefetch next timestep's x rows (global_prefetch_b8 path)
    if (s + 1 < T) {
      const int tn = dir ? (T - 2 - s) : (s + 1);
      __builtin_prefetch(x + ((size_t)arow * T + tn) * DIN, 0, 1);
    }

    __threadfence();
    __syncthreads();

    // ---- inter-WGP barrier (32 blocks per direction), monotonic counter
    if (tid == 0) {
      __hip_atomic_fetch_add(&barrier_cnt[dir], 1, __ATOMIC_ACQ_REL,
                             __HIP_MEMORY_SCOPE_AGENT);
      const int target = 32 * (s + 1);
      while (__hip_atomic_load(&barrier_cnt[dir], __ATOMIC_ACQUIRE,
                               __HIP_MEMORY_SCOPE_AGENT) < target)
        __builtin_amdgcn_s_sleep(1);
    }
    __syncthreads();
    __threadfence();
  }
}

// ---------------------------------------------------------------------------
// Small dense GEMM + bias (tail ops; A may be f16 or f32, B/bias f32).
// ---------------------------------------------------------------------------
__global__ void gemm_bias_kernel(const void* __restrict__ A, int aHalf,
                                 const float* __restrict__ B,
                                 const float* __restrict__ bias,
                                 float* __restrict__ C, int M, int N, int K) {
  int idx = blockIdx.x * blockDim.x + threadIdx.x;
  if (idx >= M * N) return;
  int mrow = idx / N, n = idx - mrow * N;
  float acc = bias ? bias[n] : 0.f;
  if (aHalf) {
    const _Float16* Ah = (const _Float16*)A;
    for (int k = 0; k < K; ++k) acc += (float)Ah[(size_t)mrow * K + k] * B[(size_t)k * N + n];
  } else {
    const float* Af = (const float*)A;
    for (int k = 0; k < K; ++k) acc += Af[(size_t)mrow * K + k] * B[(size_t)k * N + n];
  }
  C[idx] = acc;
}

// ---------------------------------------------------------------------------
// Memory retrieval (softmax ext mem + 5 cosine top-5 banks) + MLP tail.
// One block (64 threads) per batch row.
// ---------------------------------------------------------------------------
__global__ void memory_tail_kernel(
    const float* __restrict__ qbuf,
    const float* __restrict__ ext_keys, const float* __restrict__ ext_vals,
    const float* __restrict__ form_emb,
    const float* __restrict__ conc_keys, const float* __restrict__ conc_vals,
    const float* __restrict__ st_mem,
    const float* __restrict__ lt_mem, const float* __restrict__ lt_imp,
    const float* __restrict__ inf_mem, const float* __restrict__ inf_conf,
    const float* __restrict__ rs_W, const float* __restrict__ rs_b,
    const float* __restrict__ bn_gamma, const float* __restrict__ bn_beta,
    const float* __restrict__ bn_mean, const float* __restrict__ bn_var,
    const float* __restrict__ out_W, const float* __restrict__ out_b,
    const float* __restrict__ fin_W, const float* __restrict__ fin_b,
    float* __restrict__ out) {
  __shared__ float sQ[64], sQn[64], sAcc[64], sTmp[512], sX5[128];
  __shared__ float sCand[320];
  __shared__ int   sCandI[320];
  __shared__ int   sSel[5];
  __shared__ float sRed[2];

  const int b = blockIdx.x, tid = threadIdx.x; // 64 threads
  sQ[tid]  = qbuf[b * 64 + tid];
  sAcc[tid] = 0.f;
  __syncthreads();
  if (tid == 0) {
    float ss = 0.f;
    for (int e = 0; e < 64; ++e) ss += sQ[e] * sQ[e];
    sRed[0] = sqrtf(ss) + 1e-6f;
  }
  __syncthreads();
  sQn[tid] = sQ[tid] / sRed[0];
  __syncthreads();

  // external memory: softmax(q @ K^T) @ V  (raw q, not normalized)
  for (int n = tid; n < 100; n += 64) {
    float d = 0.f;
    for (int e = 0; e < 64; ++e) d += sQ[e] * ext_keys[n * 64 + e];
    sTmp[n] = d;
  }
  __syncthreads();
  if (tid == 0) {
    float mx = -1e30f;
    for (int n = 0; n < 100; ++n) mx = fmaxf(mx, sTmp[n]);
    float sum = 0.f;
    for (int n = 0; n < 100; ++n) sum += expf(sTmp[n] - mx);
    sRed[0] = mx; sRed[1] = sum;
  }
  __syncthreads();
  {
    float a = 0.f;
    for (int n = 0; n < 100; ++n)
      a += (expf(sTmp[n] - sRed[0]) / sRed[1]) * ext_vals[n * 64 + tid];
    sAcc[tid] += a;
  }
  __syncthreads();

  // 5 cosine-similarity top-5 banks
  const float* bk_keys[5] = {form_emb, conc_keys, st_mem, lt_mem, inf_mem};
  const float* bk_vals[5] = {form_emb, conc_vals, st_mem, lt_mem, inf_mem};
  const float* bk_w[5]    = {nullptr, nullptr, nullptr, lt_imp, inf_conf};
  const int    bk_n[5]    = {1000, 1000, 100, 10000, 500};
  for (int bk = 0; bk < 5; ++bk) {
    float tv[5]; int ti[5];
#pragma unroll
    for (int j = 0; j < 5; ++j) { tv[j] = -1e30f; ti[j] = 0; }
    const float* keys = bk_keys[bk];
    const float* w    = bk_w[bk];
    for (int n = tid; n < bk_n[bk]; n += 64) {
      float d = 0.f, k2 = 0.f;
      const float* kr = keys + (size_t)n * 64;
      for (int e = 0; e < 64; ++e) { float kv = kr[e]; d += sQn[e] * kv; k2 += kv * kv; }
      float sim = d / (sqrtf(k2) + 1e-6f);
      if (w) sim *= w[n];
      if (sim > tv[4]) {
        int j = 4;
        while (j > 0 && sim > tv[j - 1]) { tv[j] = tv[j - 1]; ti[j] = ti[j - 1]; --j; }
        tv[j] = sim; ti[j] = n;
      }
    }
#pragma unroll
    for (int j = 0; j < 5; ++j) { sCand[tid * 5 + j] = tv[j]; sCandI[tid * 5 + j] = ti[j]; }
    __syncthreads();
    if (tid == 0) {
      for (int pick = 0; pick < 5; ++pick) {
        int best = 0; float bv = -1e30f;
        for (int c = 0; c < 320; ++c)
          if (sCand[c] > bv) { bv = sCand[c]; best = c; }
        sSel[pick] = sCandI[best];
        sCand[best] = -2e30f;
      }
    }
    __syncthreads();
    {
      const float* vals = bk_vals[bk];
      float a = 0.f;
      for (int j = 0; j < 5; ++j) a += vals[(size_t)sSel[j] * 64 + tid];
      sAcc[tid] += a;
    }
    __syncthreads();
  }

  // mean over 26 memory slots (1 ext + 5 banks x top-5)
  sAcc[tid] *= (1.f / 26.f);
  __syncthreads();

  // relu(x @ rs_W + rs_b) then batchnorm -> sTmp[512]
  for (int j = tid; j < 512; j += 64) {
    float v = rs_b[j];
    for (int e = 0; e < 64; ++e) v += sAcc[e] * rs_W[e * 512 + j];
    v = fmaxf(v, 0.f);
    v = (v - bn_mean[j]) * rsqrtf(bn_var[j] + 1e-3f) * bn_gamma[j] + bn_beta[j];
    sTmp[j] = v;
  }
  __syncthreads();
  for (int j = tid; j < 128; j += 64) {
    float v = out_b[j];
    for (int k = 0; k < 512; ++k) v += sTmp[k] * out_W[k * 128 + j];
    sX5[j] = v;
  }
  __syncthreads();
  if (tid == 0) {
    float l = fin_b[0];
    for (int k = 0; k < 128; ++k) l += sX5[k] * fin_W[k];
    out[b] = 1.f / (1.f + expf(-l));
  }
}

// ---------------------------------------------------------------------------
extern "C" void kernel_launch(void* const* d_in, const int* in_sizes, int n_in,
                              void* d_out, int out_size, void* d_ws, size_t ws_size,
                              hipStream_t stream) {
  (void)in_sizes; (void)n_in; (void)out_size; (void)ws_size;

  const int*   tokens  = (const int*)d_in[0];
  const float* emb     = (const float*)d_in[1];
  const float* l1f_Wx  = (const float*)d_in[2];
  const float* l1f_Wh  = (const float*)d_in[3];
  const float* l1f_b   = (const float*)d_in[4];
  const float* l1b_Wx  = (const float*)d_in[5];
  const float* l1b_Wh  = (const float*)d_in[6];
  const float* l1b_b   = (const float*)d_in[7];
  const float* l2f_Wx  = (const float*)d_in[8];
  const float* l2f_Wh  = (const float*)d_in[9];
  const float* l2f_b   = (const float*)d_in[10];
  const float* l2b_Wx  = (const float*)d_in[11];
  const float* l2b_Wh  = (const float*)d_in[12];
  const float* l2b_b   = (const float*)d_in[13];
  // d_in[14..17] = Wq,bq,Wk,bk: dead (softmax over a single key == 1)
  const float* Wv      = (const float*)d_in[18];
  const float* bv      = (const float*)d_in[19];
  const float* Wo      = (const float*)d_in[20];
  const float* bo      = (const float*)d_in[21];
  const float* mq_W    = (const float*)d_in[22];
  const float* mq_b    = (const float*)d_in[23];
  const float* ext_keys = (const float*)d_in[24];
  const float* ext_vals = (const float*)d_in[25];
  const float* form_emb = (const float*)d_in[26];
  const float* conc_keys = (const float*)d_in[27];
  const float* conc_vals = (const float*)d_in[28];
  const float* st_mem   = (const float*)d_in[29];
  const float* lt_mem   = (const float*)d_in[30];
  const float* lt_imp   = (const float*)d_in[31];
  const float* inf_mem  = (const float*)d_in[32];
  const float* inf_conf = (const float*)d_in[33];
  const float* rs_W = (const float*)d_in[34];
  const float* rs_b = (const float*)d_in[35];
  const float* bn_g = (const float*)d_in[36];
  const float* bn_b = (const float*)d_in[37];
  const float* bn_m = (const float*)d_in[38];
  const float* bn_v = (const float*)d_in[39];
  const float* out_W = (const float*)d_in[40];
  const float* out_b = (const float*)d_in[41];
  const float* fin_W = (const float*)d_in[42];
  const float* fin_b = (const float*)d_in[43];

  uint8_t* ws = (uint8_t*)d_ws;
  size_t off = 0;
  auto alloc = [&](size_t bytes) {
    size_t p = off;
    off += (bytes + 255) & ~(size_t)255;
    return p;
  };
  const int NKB1 = (64 + 512) / 32;     // 18
  const int NKB2 = (1024 + 512) / 32;   // 48
  const int WSW1 = 2 * 32 * NKB1 * 2048;
  const int WSW2 = 2 * 32 * NKB2 * 2048;

  _Float16* x0    = (_Float16*)(ws + alloc(64ull * 512 * 64 * 2));    // embedded input f16
  _Float16* y1    = (_Float16*)(ws + alloc(64ull * 512 * 1024 * 2));  // layer-1 seq out f16
  _Float16* wsw1  = (_Float16*)(ws + alloc((size_t)WSW1 * 2));        // swizzled L1 weights
  _Float16* wsw2  = (_Float16*)(ws + alloc((size_t)WSW2 * 2));        // swizzled L2 weights
  _Float16* hbuf  = (_Float16*)(ws + alloc(2ull * 64 * 512 * 2));     // h state (2 dirs)
  float*    cbuf  = (float*)   (ws + alloc(2ull * 64 * 512 * 4));     // c state
  _Float16* x2    = (_Float16*)(ws + alloc(64ull * 1024 * 2));        // layer-2 last out
  float*    vbuf  = (float*)   (ws + alloc(64ull * 512 * 4));
  float*    x3    = (float*)   (ws + alloc(64ull * 1024 * 4));
  float*    qb    = (float*)   (ws + alloc(64ull * 64 * 4));
  int*      barrier = (int*)   (ws + alloc(256));

  // 0) weight pre-swizzle (f32 -> f16, WMMA B lane order)
  swizzle_weights_kernel<<<(WSW1 + 255) / 256, 256, 0, stream>>>(
      l1f_Wx, l1f_Wh, l1b_Wx, l1b_Wh, wsw1, 64, WSW1);
  swizzle_weights_kernel<<<(WSW2 + 255) / 256, 256, 0, stream>>>(
      l2f_Wx, l2f_Wh, l2b_Wx, l2b_Wh, wsw2, 1024, WSW2);

  // 1) embedding gather
  embed_f16_kernel<<<(64 * 512 * 64 + 255) / 256, 256, 0, stream>>>(
      tokens, emb, x0, 64 * 512 * 64);

  // 2) BiLSTM layer 1 (seq output)
  hipMemsetAsync(hbuf, 0, 2ull * 64 * 512 * 2, stream);
  hipMemsetAsync(cbuf, 0, 2ull * 64 * 512 * 4, stream);
  hipMemsetAsync(barrier, 0, 256, stream);
  size_t lds1 = (size_t)NKB1 * 2048 * 2 + 4 * 64 * 16 * 4;   // 90112 B
  hipFuncSetAttribute(reinterpret_cast<const void*>(lstm_layer_kernel<64>),
                      hipFuncAttributeMaxDynamicSharedMemorySize, (int)lds1);
  lstm_layer_kernel<64><<<64, 256, lds1, stream>>>(
      x0, wsw1, l1f_b, l1b_b, hbuf, cbuf, y1, 512, 1, barrier);

  // 3) BiLSTM layer 2 (last hidden only -> x2 [64,1024])
  hipMemsetAsync(hbuf, 0, 2ull * 64 * 512 * 2, stream);
  hipMemsetAsync(cbuf, 0, 2ull * 64 * 512 * 4, stream);
  hipMemsetAsync(barrier, 0, 256, stream);
  size_t lds2 = (size_t)NKB2 * 2048 * 2 + 4 * 64 * 16 * 4;   // 212992 B
  hipFuncSetAttribute(reinterpret_cast<const void*>(lstm_layer_kernel<1024>),
                      hipFuncAttributeMaxDynamicSharedMemorySize, (int)lds2);
  lstm_layer_kernel<1024><<<64, 256, lds2, stream>>>(
      y1, wsw2, l2f_b, l2b_b, hbuf, cbuf, x2, 512, 0, barrier);

  // 4) attention (seq len 1): v = x2@Wv+bv ; x3 = v@Wo+bo ; query = x3@mq_W+mq_b
  gemm_bias_kernel<<<(64 * 512 + 255) / 256, 256, 0, stream>>>(
      x2, 1, Wv, bv, vbuf, 64, 512, 1024);
  gemm_bias_kernel<<<(64 * 1024 + 255) / 256, 256, 0, stream>>>(
      vbuf, 0, Wo, bo, x3, 64, 1024, 512);
  gemm_bias_kernel<<<(64 * 64 + 255) / 256, 256, 0, stream>>>(
      x3, 0, mq_W, mq_b, qb, 64, 64, 1024);

  // 5) memory retrieval + MLP tail -> sigmoid [64]
  memory_tail_kernel<<<64, 64, 0, stream>>>(
      qb, ext_keys, ext_vals, form_emb, conc_keys, conc_vals, st_mem,
      lt_mem, lt_imp, inf_mem, inf_conf, rs_W, rs_b, bn_g, bn_b, bn_m, bn_v,
      out_W, out_b, fin_W, fin_b, (float*)d_out);
}
